// HierarchicalAttention_17927193493968
// MI455X (gfx1250) — matrix operand order repaired and checked by reading, not compiled, not run on hardware
//
#include <hip/hip_runtime.h>

// ---------------------------------------------------------------------------
// Types for CDNA5 WMMA (wave32): v_wmma_f32_16x16x32_bf16
// ---------------------------------------------------------------------------
typedef __attribute__((ext_vector_type(16))) __bf16 v16bf;
typedef __attribute__((ext_vector_type(8)))  float  v8f;

#define DMODEL 256
#define NHEADS 4
#define DHEAD  64
#define SEG    10
#define BATCH  16
#define SEQ    8000
#define NSEG   800                 // segments per batch
#define NSEG_TOT (BATCH * NSEG)    // 12800

// ---------------------------------------------------------------------------
// bf16 helpers (bit-level, round-to-nearest-even)
// ---------------------------------------------------------------------------
static __device__ __forceinline__ unsigned short f32_to_bf16(float f) {
  unsigned int u = __builtin_bit_cast(unsigned int, f);
  u += 0x7FFFu + ((u >> 16) & 1u);
  return (unsigned short)(u >> 16);
}
static __device__ __forceinline__ float bf16_to_f32(unsigned short s) {
  unsigned int u = ((unsigned int)s) << 16;
  return __builtin_bit_cast(float, u);
}
static __device__ __forceinline__ __bf16 us_to_bf(unsigned short s) {
  return __builtin_bit_cast(__bf16, s);
}

// ---------------------------------------------------------------------------
// Fragment loaders (16-bit A 16x32, B 32x16, f32 C 16x16), contiguous k order in A and B
//
// A (16x32, row-major source, lda elems):
//   lane L<16  : M=L, elems j = K k0+j,          j=0..15
//   lane L>=16 : M=L-16, elems j = K k0+16+j,    j=0..15
// B (32x16) built from W stored [N,K] row-major (computes A @ W^T):
//   lane L<16  : N=L, elems j = W[n0+L][k0+j],    j=0..15
//   lane L>=16 : N=L-16, elems j = W[n0+L-16][k0+16+j]
// C/D f32: VGPR i, lane<16 -> D[m0+i][n0+lane]; lane>=16 -> D[m0+i+8][n0+lane-16]
// ---------------------------------------------------------------------------
static __device__ __forceinline__ v16bf
load_a_bf16(const unsigned short* A, int lda, int row0, int k0, int lane) {
  const int half = lane >> 4, r = lane & 15;
  const unsigned short* p = A + (size_t)(row0 + r) * lda;
  const int kb1 = k0 + half * 16, kb2 = k0 + half * 16 + 8;
  v16bf f;
#pragma unroll
  for (int j = 0; j < 8; ++j) f[j]     = us_to_bf(p[kb1 + j]);
#pragma unroll
  for (int j = 0; j < 8; ++j) f[8 + j] = us_to_bf(p[kb2 + j]);
  return f;
}

// A built from fp32 source with on-the-fly bf16 convert; rows >= nrows are zero.
static __device__ __forceinline__ v16bf
load_a_f32_pad(const float* A, int lda, int nrows, int k0, int lane) {
  const int half = lane >> 4, r = lane & 15;
  v16bf f;
  if (r < nrows) {
    const float* p = A + (size_t)r * lda;
    const int kb1 = k0 + half * 16, kb2 = k0 + half * 16 + 8;
#pragma unroll
    for (int j = 0; j < 8; ++j) f[j]     = us_to_bf(f32_to_bf16(p[kb1 + j]));
#pragma unroll
    for (int j = 0; j < 8; ++j) f[8 + j] = us_to_bf(f32_to_bf16(p[kb2 + j]));
  } else {
#pragma unroll
    for (int j = 0; j < 16; ++j) f[j] = us_to_bf((unsigned short)0);
  }
  return f;
}

// B from W[N,K] row-major (A @ W^T pattern); contiguous 16-element runs.
static __device__ __forceinline__ v16bf
load_b_wt(const unsigned short* W, int ldw, int n0, int k0, int lane) {
  const int half = lane >> 4, r = lane & 15;
  const unsigned short* p = W + (size_t)(n0 + r) * ldw + k0 + half * 16;
  v16bf f;
#pragma unroll
  for (int j = 0; j < 16; ++j) f[j] = us_to_bf(p[j]);
  return f;
}

// B from a true [K,N] row-major matrix (strided column gather), for attn @ V.
static __device__ __forceinline__ v16bf
load_b_rowmajor(const unsigned short* B, int ldb, int k0, int n0, int lane) {
  const int half = lane >> 4, r = lane & 15;
  const unsigned short* p = B + (size_t)(k0 + half * 16) * ldb + n0 + r;
  v16bf f;
#pragma unroll
  for (int j = 0; j < 16; ++j) f[j] = us_to_bf(p[(size_t)j * ldb]);
  return f;
}

// ---------------------------------------------------------------------------
// fp32 -> bf16 array conversion
// ---------------------------------------------------------------------------
__global__ void k_convert_bf16(const float* __restrict__ src,
                               unsigned short* __restrict__ dst, int n) {
  for (int i = blockIdx.x * blockDim.x + threadIdx.x; i < n;
       i += gridDim.x * blockDim.x)
    dst[i] = f32_to_bf16(src[i]);
}

// ---------------------------------------------------------------------------
// Generic GEMM: C[M,N] = A[M,K](bf16) @ W[N,K]^T(bf16) + bias[N]
// One wave per 16x16 output tile; 8 waves per block.
// ---------------------------------------------------------------------------
__global__ void k_gemm_bias(const unsigned short* __restrict__ A,
                            const unsigned short* __restrict__ W,
                            const float* __restrict__ bias,
                            float* __restrict__ Cf,          // optional
                            unsigned short* __restrict__ Cb, // optional
                            int M, int N, int K) {
  const int lane = threadIdx.x & 31;
  const int wave = threadIdx.x >> 5;
  const int ntn  = N >> 4;
  const int gid  = blockIdx.x * (blockDim.x >> 5) + wave;
  const int total = (M >> 4) * ntn;
  if (gid >= total) return;              // whole wave exits together
  const int tm = gid / ntn, tn = gid % ntn;
  const int row0 = tm << 4, n0 = tn << 4;

  v8f c = {};
  for (int k0 = 0; k0 < K; k0 += 32) {
    v16bf a = load_a_bf16(A, K, row0, k0, lane);
    v16bf b = load_b_wt(W, K, n0, k0, lane);
    c = __builtin_amdgcn_wmma_f32_16x16x32_bf16(false, a, false, b,
                                                (short)0, c, false, false);
  }
  const int half = lane >> 4;
  const int cn = n0 + (lane & 15);
  const float bi = bias ? bias[cn] : 0.0f;
#pragma unroll
  for (int i = 0; i < 8; ++i) {
    const int m = row0 + i + half * 8;
    const float v = c[i] + bi;
    const size_t idx = (size_t)m * N + cn;
    if (Cf) Cf[idx] = v;
    if (Cb) Cb[idx] = f32_to_bf16(v);
  }
}

// ---------------------------------------------------------------------------
// Intra-segment: fused QKV (WMMA) + 10x10 MHA (VALU) + weight output +
// column-mean of attention output (pre-Wo pooling, linearity exploited).
// One block (256 thr = 8 waves) per segment.
// ---------------------------------------------------------------------------
__global__ void k_intra(const float* __restrict__ x,
                        const unsigned short* __restrict__ Wqkvb, // [768,256]
                        const float* __restrict__ bqkv,
                        float* __restrict__ intra_w,   // [NSEG_TOT,10,10]
                        unsigned short* __restrict__ mavg) { // [NSEG_TOT,256]
  __shared__ float qkv[SEG][3 * DMODEL];   // 30720 B
  __shared__ float sc[NHEADS][SEG][SEG];   //  1600 B
  __shared__ float ao[SEG][DMODEL];        // 10240 B

  const int seg  = blockIdx.x;
  const int lane = threadIdx.x & 31;
  const int wave = threadIdx.x >> 5;
  const float* xseg = x + (size_t)seg * SEG * DMODEL;

  // ---- QKV = x_seg @ Wqkv^T + bqkv : 48 N-tiles of 16, K = 256 ----
  for (int nt = wave; nt < 48; nt += 8) {
    const int n0 = nt << 4;
    v8f c = {};
    for (int k0 = 0; k0 < DMODEL; k0 += 32) {
      v16bf a = load_a_f32_pad(xseg, DMODEL, SEG, k0, lane);
      v16bf b = load_b_wt(Wqkvb, DMODEL, n0, k0, lane);
      c = __builtin_amdgcn_wmma_f32_16x16x32_bf16(false, a, false, b,
                                                  (short)0, c, false, false);
    }
    const int half = lane >> 4;
    const int cn = n0 + (lane & 15);
    const float bi = bqkv[cn];
#pragma unroll
    for (int i = 0; i < 8; ++i) {
      const int m = i + half * 8;
      if (m < SEG) qkv[m][cn] = c[i] + bi;
    }
  }
  __syncthreads();

  // ---- scores = q k^T / 8 ----
  for (int t = threadIdx.x; t < NHEADS * SEG * SEG; t += blockDim.x) {
    const int h = t / (SEG * SEG), rem = t % (SEG * SEG);
    const int qi = rem / SEG, ki = rem % SEG;
    const float* qp = &qkv[qi][h * DHEAD];
    const float* kp = &qkv[ki][DMODEL + h * DHEAD];
    float s = 0.f;
#pragma unroll 8
    for (int d = 0; d < DHEAD; ++d) s += qp[d] * kp[d];
    sc[h][qi][ki] = s * 0.125f;
  }
  __syncthreads();

  // ---- softmax over ki ----
  for (int t = threadIdx.x; t < NHEADS * SEG; t += blockDim.x) {
    const int h = t / SEG, qi = t % SEG;
    float mx = sc[h][qi][0];
    for (int k = 1; k < SEG; ++k) mx = fmaxf(mx, sc[h][qi][k]);
    float sum = 0.f;
    for (int k = 0; k < SEG; ++k) {
      const float e = __expf(sc[h][qi][k] - mx);
      sc[h][qi][k] = e; sum += e;
    }
    const float inv = 1.0f / sum;
    for (int k = 0; k < SEG; ++k) sc[h][qi][k] *= inv;
  }
  __syncthreads();

  // ---- intra attention weights (mean over heads) ----
  for (int t = threadIdx.x; t < SEG * SEG; t += blockDim.x) {
    const int qi = t / SEG, ki = t % SEG;
    intra_w[(size_t)seg * SEG * SEG + t] =
        0.25f * (sc[0][qi][ki] + sc[1][qi][ki] + sc[2][qi][ki] + sc[3][qi][ki]);
  }

  // ---- attn @ v ----
  for (int t = threadIdx.x; t < SEG * DMODEL; t += blockDim.x) {
    const int qi = t / DMODEL, col = t % DMODEL, h = col / DHEAD;
    float s = 0.f;
#pragma unroll
    for (int k = 0; k < SEG; ++k) s += sc[h][qi][k] * qkv[k][2 * DMODEL + col];
    ao[qi][col] = s;
  }
  __syncthreads();

  // ---- column mean over the 10 rows (Wo projection commutes with mean) ----
  for (int d = threadIdx.x; d < DMODEL; d += blockDim.x) {
    float s = 0.f;
#pragma unroll
    for (int r = 0; r < SEG; ++r) s += ao[r][d];
    mavg[(size_t)seg * DMODEL + d] = f32_to_bf16(s * 0.1f);
  }
}

// ---------------------------------------------------------------------------
// Inter attention: scores + softmax per (batch, head, 16-query tile).
// qkv2b: bf16 [NSEG_TOT, 768] = [q|k|v]. Writes bf16 attn to ws.
// ---------------------------------------------------------------------------
__global__ void k_inter_scores(const unsigned short* __restrict__ qkv2b,
                               unsigned short* __restrict__ attnb) {
  __shared__ float sc[16][NSEG];           // 51200 B
  const int qt = blockIdx.x;               // 0..49
  const int h  = blockIdx.y;               // 0..3
  const int b  = blockIdx.z;               // 0..15
  const int lane = threadIdx.x & 31;
  const int wave = threadIdx.x >> 5;

  const unsigned short* qbase =
      qkv2b + (size_t)(b * NSEG + qt * 16) * 768 + h * DHEAD;

  // scores[qi][kcol] = q . k / 8 via WMMA, K = 64 (2 steps)
  for (int kt = wave; kt < NSEG / 16; kt += 8) {
    const unsigned short* kbase =
        qkv2b + (size_t)(b * NSEG + kt * 16) * 768 + DMODEL + h * DHEAD;
    v8f c = {};
    for (int k0 = 0; k0 < DHEAD; k0 += 32) {
      v16bf a = load_a_bf16(qbase, 768, 0, k0, lane);
      v16bf bb = load_b_wt(kbase, 768, 0, k0, lane);
      c = __builtin_amdgcn_wmma_f32_16x16x32_bf16(false, a, false, bb,
                                                  (short)0, c, false, false);
    }
    const int half = lane >> 4;
    const int kc = kt * 16 + (lane & 15);
#pragma unroll
    for (int i = 0; i < 8; ++i) sc[i + half * 8][kc] = c[i] * 0.125f;
  }
  __syncthreads();

  // full-row softmax (needed anyway: attn weights are an output)
  if (threadIdx.x < 16) {
    const int r = threadIdx.x;
    float mx = sc[r][0];
    for (int k = 1; k < NSEG; ++k) mx = fmaxf(mx, sc[r][k]);
    float sum = 0.f;
    for (int k = 0; k < NSEG; ++k) {
      const float e = __expf(sc[r][k] - mx);
      sc[r][k] = e; sum += e;
    }
    const float inv = 1.0f / sum;
    for (int k = 0; k < NSEG; ++k) sc[r][k] *= inv;
  }
  __syncthreads();

  // write bf16 attn tile
  unsigned short* dst =
      attnb + ((size_t)(b * NHEADS + h) * NSEG + qt * 16) * NSEG;
  for (int idx = threadIdx.x; idx < 16 * NSEG; idx += blockDim.x) {
    const int r = idx / NSEG, col = idx % NSEG;
    dst[(size_t)r * NSEG + col] = f32_to_bf16(sc[r][col]);
  }
}

// inter attention weights: mean over heads
__global__ void k_head_mean(const unsigned short* __restrict__ attnb,
                            float* __restrict__ inter_w) {
  const size_t n = (size_t)BATCH * NSEG * NSEG;
  for (size_t i = blockIdx.x * (size_t)blockDim.x + threadIdx.x; i < n;
       i += (size_t)gridDim.x * blockDim.x) {
    const int b = (int)(i / ((size_t)NSEG * NSEG));
    const size_t rem = i % ((size_t)NSEG * NSEG);
    float s = 0.f;
#pragma unroll
    for (int h = 0; h < NHEADS; ++h)
      s += bf16_to_f32(attnb[((size_t)(b * NHEADS + h) * NSEG * NSEG) + rem]);
    inter_w[i] = 0.25f * s;
  }
}

// out[16 q-rows, 64 head-dims] = attn @ v  (WMMA, K = 800)
__global__ void k_attn_v(const unsigned short* __restrict__ attnb,
                         const unsigned short* __restrict__ qkv2b,
                         unsigned short* __restrict__ ao2b) { // [NSEG_TOT,256]
  const int qt = blockIdx.x, h = blockIdx.y, b = blockIdx.z;
  const int lane = threadIdx.x & 31;
  const int wave = threadIdx.x >> 5;          // 4 waves, one 16-col tile each
  const int n0 = wave * 16;

  const unsigned short* A =
      attnb + ((size_t)(b * NHEADS + h) * NSEG + qt * 16) * NSEG;
  const unsigned short* V =
      qkv2b + (size_t)(b * NSEG) * 768 + 2 * DMODEL + h * DHEAD;

  v8f c = {};
  for (int k0 = 0; k0 < NSEG; k0 += 32) {
    v16bf a = load_a_bf16(A, NSEG, 0, k0, lane);
    v16bf bb = load_b_rowmajor(V, 768, k0, n0, lane);
    c = __builtin_amdgcn_wmma_f32_16x16x32_bf16(false, a, false, bb,
                                                (short)0, c, false, false);
  }
  const int half = lane >> 4;
  const int col = h * DHEAD + n0 + (lane & 15);
#pragma unroll
  for (int i = 0; i < 8; ++i) {
    const int m = qt * 16 + i + half * 8;
    ao2b[(size_t)(b * NSEG + m) * DMODEL + col] = f32_to_bf16(c[i]);
  }
}

// broadcast the per-segment projected output to all 10 rows of each segment
__global__ void k_broadcast(const float* __restrict__ fin, // [NSEG_TOT,256]
                            float* __restrict__ out) {     // [B*SEQ,256]
  const size_t n = (size_t)BATCH * SEQ * DMODEL;
  for (size_t i = blockIdx.x * (size_t)blockDim.x + threadIdx.x; i < n;
       i += (size_t)gridDim.x * blockDim.x) {
    const size_t row = i >> 8;        // /256
    const int col = (int)(i & 255);
    out[i] = fin[(row / SEG) * DMODEL + col];
  }
}

// ---------------------------------------------------------------------------
// Host launcher
// ---------------------------------------------------------------------------
static inline size_t align256(size_t x) { return (x + 255) & ~(size_t)255; }

extern "C" void kernel_launch(void* const* d_in, const int* in_sizes, int n_in,
                              void* d_out, int out_size, void* d_ws, size_t ws_size,
                              hipStream_t stream) {
  (void)in_sizes; (void)n_in; (void)out_size; (void)ws_size;
  const float* x          = (const float*)d_in[0];
  const float* intra_Wqkv = (const float*)d_in[1];
  const float* intra_bqkv = (const float*)d_in[2];
  const float* intra_Wo   = (const float*)d_in[3];
  const float* intra_bo   = (const float*)d_in[4];
  const float* seg_W      = (const float*)d_in[5];
  const float* seg_b      = (const float*)d_in[6];
  const float* inter_Wqkv = (const float*)d_in[7];
  const float* inter_bqkv = (const float*)d_in[8];
  const float* inter_Wo   = (const float*)d_in[9];
  const float* inter_bo   = (const float*)d_in[10];
  const float* out_W      = (const float*)d_in[11];
  const float* out_b      = (const float*)d_in[12];

  float* out_main  = (float*)d_out;                                  // [16,8000,256]
  float* out_intra = out_main + (size_t)BATCH * SEQ * DMODEL;        // [16,800,10,10]
  float* out_inter = out_intra + (size_t)BATCH * NSEG * SEG * SEG;   // [16,800,800]

  // ---- workspace carve (bf16 staging + intermediates), ~148 MB ----
  char* p = (char*)d_ws;
  auto carve = [&](size_t bytes) { char* q = p; p += align256(bytes); return q; };
  unsigned short* Wqkv1b = (unsigned short*)carve(768 * 256 * 2);
  unsigned short* Wo1b   = (unsigned short*)carve(256 * 256 * 2);
  unsigned short* segWb  = (unsigned short*)carve(256 * 256 * 2);
  unsigned short* Wqkv2b = (unsigned short*)carve(768 * 256 * 2);
  unsigned short* Wo2b   = (unsigned short*)carve(256 * 256 * 2);
  unsigned short* outWb  = (unsigned short*)carve(256 * 256 * 2);
  unsigned short* mavg   = (unsigned short*)carve((size_t)NSEG_TOT * 256 * 2);
  unsigned short* t1b    = (unsigned short*)carve((size_t)NSEG_TOT * 256 * 2);
  unsigned short* repsB  = (unsigned short*)carve((size_t)NSEG_TOT * 256 * 2);
  unsigned short* qkv2b  = (unsigned short*)carve((size_t)NSEG_TOT * 768 * 2);
  unsigned short* attnb  = (unsigned short*)carve((size_t)BATCH * NHEADS * NSEG * NSEG * 2);
  unsigned short* ao2b   = (unsigned short*)carve((size_t)NSEG_TOT * 256 * 2);
  unsigned short* t2b    = (unsigned short*)carve((size_t)NSEG_TOT * 256 * 2);
  float*          fin    = (float*)carve((size_t)NSEG_TOT * 256 * 4);

  // ---- stage weights as bf16 ----
  k_convert_bf16<<<768, 256, 0, stream>>>(intra_Wqkv, Wqkv1b, 768 * 256);
  k_convert_bf16<<<256, 256, 0, stream>>>(intra_Wo,   Wo1b,   256 * 256);
  k_convert_bf16<<<256, 256, 0, stream>>>(seg_W,      segWb,  256 * 256);
  k_convert_bf16<<<768, 256, 0, stream>>>(inter_Wqkv, Wqkv2b, 768 * 256);
  k_convert_bf16<<<256, 256, 0, stream>>>(inter_Wo,   Wo2b,   256 * 256);
  k_convert_bf16<<<256, 256, 0, stream>>>(out_W,      outWb,  256 * 256);

  // ---- intra attention (fused, one block per segment) ----
  k_intra<<<NSEG_TOT, 256, 0, stream>>>(x, Wqkv1b, intra_bqkv, out_intra, mavg);

  auto gemm = [&](const unsigned short* A, const unsigned short* W,
                  const float* bias, float* Cf, unsigned short* Cb,
                  int M, int N, int K) {
    const int tiles = (M / 16) * (N / 16);
    k_gemm_bias<<<(tiles + 7) / 8, 256, 0, stream>>>(A, W, bias, Cf, Cb, M, N, K);
  };

  // mean(x_intra) @ Wo^T + bo  (pooling commuted before Wo)
  gemm(mavg, Wo1b, intra_bo, nullptr, t1b, NSEG_TOT, 256, 256);
  // segment_reps = t1 @ seg_W^T + seg_b
  gemm(t1b, segWb, seg_b, nullptr, repsB, NSEG_TOT, 256, 256);
  // inter QKV
  gemm(repsB, Wqkv2b, inter_bqkv, nullptr, qkv2b, NSEG_TOT, 768, 256);

  // inter attention
  k_inter_scores<<<dim3(NSEG / 16, NHEADS, BATCH), 256, 0, stream>>>(qkv2b, attnb);
  {
    const size_t n = (size_t)BATCH * NSEG * NSEG;
    k_head_mean<<<(unsigned)((n + 255) / 256), 256, 0, stream>>>(attnb, out_inter);
  }
  k_attn_v<<<dim3(NSEG / 16, NHEADS, BATCH), 128, 0, stream>>>(attnb, qkv2b, ao2b);

  // inter out-proj, then final projection (commuted before broadcast)
  gemm(ao2b, Wo2b, inter_bo, nullptr, t2b, NSEG_TOT, 256, 256);
  gemm(t2b, outWb, out_b, fin, nullptr, NSEG_TOT, 256, 256);

  // broadcast to full sequence
  {
    const size_t n = (size_t)BATCH * SEQ * DMODEL;
    k_broadcast<<<(unsigned)((n + 255) / 256), 256, 0, stream>>>(fin, out_main);
  }
}
